// Block_11802570129472
// MI455X (gfx1250) — matrix operand order repaired and checked by reading, not compiled, not run on hardware
//
#include <hip/hip_runtime.h>
#include <hip/hip_bf16.h>

#define DIMSZ 1024
#define NH    16
#define NKV   4
#define HDIM  64
#define SLEN  2048
#define BATCH 4
#define MTOK  (BATCH * SLEN)

typedef __bf16 bf16;
typedef __attribute__((ext_vector_type(16))) __bf16 v16bf;
typedef __attribute__((ext_vector_type(8)))  float  v8f;

struct U128x2 { uint4 lo, hi; };

static __device__ __forceinline__ v16bf pack_ab(uint4 lo, uint4 hi) {
    U128x2 u{lo, hi};
    return __builtin_bit_cast(v16bf, u);
}

static __device__ __forceinline__ v8f wmma_bf16(v16bf a, v16bf b, v8f c) {
    // D = A(16x32) * B(32x16) + C, fp32 accumulate
    return __builtin_amdgcn_wmma_f32_16x16x32_bf16(
        /*neg_a=*/false, a, /*neg_b=*/false, b,
        /*c_mod=*/(short)0, c, /*reuse_a=*/false, /*reuse_b=*/false);
}

static __device__ __forceinline__ float redmax16(float v) {
    v = fmaxf(v, __shfl_xor(v, 8));
    v = fmaxf(v, __shfl_xor(v, 4));
    v = fmaxf(v, __shfl_xor(v, 2));
    v = fmaxf(v, __shfl_xor(v, 1));
    return v;
}
static __device__ __forceinline__ float redsum16(float v) {
    v += __shfl_xor(v, 8);
    v += __shfl_xor(v, 4);
    v += __shfl_xor(v, 2);
    v += __shfl_xor(v, 1);
    return v;
}

// One 16-byte global -> LDS async transfer per lane (ASYNCcnt-tracked).
static __device__ __forceinline__ void async_g2l_b128(unsigned lds_off, const void* gptr) {
    asm volatile("global_load_async_to_lds_b128 %0, %1, off"
                 :: "v"(lds_off), "v"((unsigned long long)(size_t)gptr)
                 : "memory");
}
static __device__ __forceinline__ void wait_async0() {
    asm volatile("s_wait_asynccnt 0x0" ::: "memory");
}

// ---------------- fp32 -> bf16 convert ----------------
__global__ __launch_bounds__(256) void tobf16(const float* __restrict__ src,
                                              bf16* __restrict__ dst, int n) {
    int i = blockIdx.x * 256 + threadIdx.x;
    if (i < n) dst[i] = (bf16)src[i];
}

// ---------------- NT GEMM: C[M,N] = A[M,K] * W[N,K]^T ----------------
// A, W row-major bf16; C fp32. Block = 4 waves, each wave = 64x64 tile (4x4 WMMA tiles).
__global__ __launch_bounds__(128) void gemm_nt_bf16(
    const bf16* __restrict__ A, const bf16* __restrict__ W,
    float* __restrict__ C, int M, int N, int K) {
    const int lane = threadIdx.x & 31;
    const int wave = threadIdx.x >> 5;
    const int ml = lane & 15;
    const int kh = lane >> 4;
    const int m0 = (blockIdx.y * 2 + (wave >> 1)) * 64;
    const int n0 = (blockIdx.x * 2 + (wave & 1)) * 64;

    v8f acc[4][4] = {};
    for (int k0 = 0; k0 < K; k0 += 32) {
        v16bf a[4], b[4];
#pragma unroll
        for (int i = 0; i < 4; ++i) {
            // A layout: lane holds row (m0+i*16+ml), K chunks [kh*8, +8) and [16+kh*8, +8)
            const bf16* ap = A + (size_t)(m0 + i * 16 + ml) * K + k0 + kh * 8;
            a[i] = pack_ab(*(const uint4*)ap, *(const uint4*)(ap + 16));
        }
#pragma unroll
        for (int j = 0; j < 4; ++j) {
            // B layout: lane holds column (n0+j*16+ml) = W row, K chunks [kh*8, +8) and [16+kh*8, +8)
            const bf16* bp = W + (size_t)(n0 + j * 16 + ml) * K + k0 + kh * 8;
            b[j] = pack_ab(*(const uint4*)bp, *(const uint4*)(bp + 16));
        }
#pragma unroll
        for (int i = 0; i < 4; ++i)
#pragma unroll
            for (int j = 0; j < 4; ++j)
                acc[i][j] = wmma_bf16(a[i], b[j], acc[i][j]);
    }
    // C layout: VGPR r <-> row r + 8*(lane>>4); col = lane&15
#pragma unroll
    for (int i = 0; i < 4; ++i)
#pragma unroll
        for (int j = 0; j < 4; ++j)
#pragma unroll
            for (int r = 0; r < 8; ++r) {
                int m = m0 + i * 16 + r + kh * 8;
                int n = n0 + j * 16 + ml;
                C[(size_t)m * N + n] = acc[i][j][r];
            }
}

// ---------------- per-head RMSNorm + RoPE + gain, fp32 -> bf16 [B,heads,S,HD] --------
__global__ __launch_bounds__(256) void normrope(
    const float* __restrict__ src,  // [B,S,heads*HD] (projection layout)
    bf16* __restrict__ dst,         // [B,heads,S,HD]
    const float* __restrict__ gain, // per-head gain or nullptr
    int heads) {
    const int lane = threadIdx.x & 31;
    int row = blockIdx.x * 8 + (threadIdx.x >> 5); // (b*heads + h)*S + s
    int s = row % SLEN;
    int h = (row / SLEN) % heads;
    int b = row / (SLEN * heads);
    const float* p = src + ((size_t)(b * SLEN + s)) * (heads * HDIM) + h * HDIM;
    float t1 = p[lane];
    float t2 = p[lane + 32];
    float ss = t1 * t1 + t2 * t2;
    ss += __shfl_xor(ss, 16);
    ss += __shfl_xor(ss, 8);
    ss += __shfl_xor(ss, 4);
    ss += __shfl_xor(ss, 2);
    ss += __shfl_xor(ss, 1);
    float scale = rsqrtf(ss * (1.0f / 64.0f) + 1e-6f);
    t1 *= scale; t2 *= scale;
    // inv_freq[i] = 10000^(-i/32) = exp(-i * ln(10000)/32)
    float ang = (float)s * __expf(-(float)lane * 0.28782313662425572f);
    float sn, cs;
    __sincosf(ang, &sn, &cs);
    float g = gain ? gain[h] : 1.0f;
    float o1 = (t1 * cs + t2 * sn) * g;
    float o2 = (-t1 * sn + t2 * cs) * g;
    bf16* q = dst + ((size_t)((b * heads + h) * SLEN + s)) * HDIM;
    q[lane] = (bf16)o1;
    q[lane + 32] = (bf16)o2;
}

// ---------------- V: fp32 [B,S,NKV*HD] -> bf16 transposed [B,NKV,HD,S] ----------------
__global__ __launch_bounds__(256) void vtrans(const float* __restrict__ v,
                                              bf16* __restrict__ vt) {
    size_t i = (size_t)blockIdx.x * 256 + threadIdx.x; // over MTOK*NKV*HD
    int c = (int)(i % (NKV * HDIM));
    size_t bs = i / (NKV * HDIM);
    int s = (int)(bs % SLEN);
    int b = (int)(bs / SLEN);
    int kvh = c >> 6, d = c & 63;
    vt[((size_t)((b * NKV + kvh) * HDIM + d)) * SLEN + s] = (bf16)v[i];
}

// ---------------- flash attention: causal GQA ----------------
// Block = 8 waves sharing one (b, kv-head); each wave owns one 16-row q tile.
// K/V blocks staged once per workgroup via async global->LDS (double buffered).
__global__ __launch_bounds__(256) void attn_fa(
    const bf16* __restrict__ Q,   // [B,NH,S,HD]
    const bf16* __restrict__ Kc,  // [B,NKV,S,HD]
    const bf16* __restrict__ Vt,  // [B,NKV,HD,S]
    bf16* __restrict__ Y) {       // [B,S,NH*HD]
    __shared__ __attribute__((aligned(16))) bf16 kbuf[2][32 * HDIM]; // 2 x 4KB
    __shared__ __attribute__((aligned(16))) bf16 vbuf[2][HDIM * 32]; // 2 x 4KB
    __shared__ __attribute__((aligned(16))) bf16 pbuf[8][16 * 32];   // 8KB
    const int tid = threadIdx.x;
    const int lane = tid & 31;
    const int wave = tid >> 5;
    const int ml = lane & 15;
    const int kh = lane >> 4;
    const int QT = SLEN / 16;
    const int wid = blockIdx.x * 8 + wave;
    const int qt = wid % QT;
    const int h = (wid / QT) % NH;
    const int b = wid / (QT * NH);
    const int kvh = h >> 2; // H/KVH = 4
    const int qt0 = (blockIdx.x * 8) % QT; // first q-tile in this block (QT % 8 == 0)

    const bf16* kbase = Kc + (size_t)(b * NKV + kvh) * SLEN * HDIM;
    const bf16* vbase = Vt + (size_t)(b * NKV + kvh) * HDIM * SLEN;

    // cooperative async stage of one 32-wide kv block into LDS buffer `buf`
    auto stage = [&](int kb, int buf) {
        const int j0 = kb * 32;
        {   // K: 32 rows x 64 bf16 = 32 x 8 segments of 16B
            int row = tid >> 3, seg = tid & 7;
            const bf16* g = kbase + (size_t)(j0 + row) * HDIM + seg * 8;
            async_g2l_b128((unsigned)(size_t)&kbuf[buf][row * HDIM + seg * 8], g);
        }
        {   // V^T: 64 rows(d) x 32 bf16 = 64 x 4 segments of 16B
            int d = tid >> 2, seg = tid & 3;
            const bf16* g = vbase + (size_t)d * SLEN + j0 + seg * 8;
            async_g2l_b128((unsigned)(size_t)&vbuf[buf][d * 32 + seg * 8], g);
        }
    };

    // q as WMMA-A registers (two K=32 chunks over HD=64)
    const bf16* qp = Q + ((size_t)((b * NH + h) * SLEN) + qt * 16 + ml) * HDIM;
    v16bf qa[2];
#pragma unroll
    for (int c = 0; c < 2; ++c) {
        const bf16* p = qp + c * 32 + kh * 8;
        qa[c] = pack_ab(*(const uint4*)p, *(const uint4*)(p + 16));
    }

    float mi[8], li[8];
#pragma unroll
    for (int r = 0; r < 8; ++r) { mi[r] = -1e30f; li[r] = 0.0f; }
    v8f oacc[4] = {};

    const int nown = (qt * 16 + 16 + 31) >> 5;        // causal trip count (this wave)
    const int nmax = ((qt0 + 7) * 16 + 16 + 31) >> 5; // block-wide trip count (uniform)

    stage(0, 0);
    for (int kb = 0; kb < nmax; ++kb) {
        const int par = kb & 1;
        wait_async0();       // my async stores into buf[par] are in LDS
        __syncthreads();     // everyone's stores done; buf[par^1] readers all finished
        if (kb + 1 < nmax) stage(kb + 1, par ^ 1); // overlap DMA with compute
        if (kb < nown) {
            const int j0 = kb * 32;
            v8f sc[2] = {};
#pragma unroll
            for (int t = 0; t < 2; ++t) {
                const bf16* kr = &kbuf[par][(t * 16 + ml) * HDIM];
#pragma unroll
                for (int c = 0; c < 2; ++c) {
                    const bf16* p = kr + c * 32 + kh * 8;
                    v16bf kbv = pack_ab(*(const uint4*)p, *(const uint4*)(p + 16));
                    sc[t] = wmma_bf16(qa[c], kbv, sc[t]);
                }
            }
            // scale 1/sqrt(HD), causal mask, online softmax
#pragma unroll
            for (int r = 0; r < 8; ++r) {
                int row = qt * 16 + r + kh * 8;
                float s0 = sc[0][r] * 0.125f;
                float s1 = sc[1][r] * 0.125f;
                if (j0 + ml > row) s0 = -1e30f;
                if (j0 + 16 + ml > row) s1 = -1e30f;
                float rmax = redmax16(fmaxf(s0, s1));
                float mnew = fmaxf(mi[r], rmax);
                float corr = __expf(mi[r] - mnew);
                mi[r] = mnew;
                float p0 = __expf(s0 - mnew);
                float p1 = __expf(s1 - mnew);
                li[r] = li[r] * corr + redsum16(p0 + p1);
#pragma unroll
                for (int dt = 0; dt < 4; ++dt) oacc[dt][r] *= corr;
                int m = r + kh * 8;
                pbuf[wave][m * 32 + ml] = (bf16)p0;
                pbuf[wave][m * 32 + 16 + ml] = (bf16)p1;
            }
            asm volatile("" ::: "memory"); // order per-wave LDS store -> load (DS in-order)
            const bf16* pp = &pbuf[wave][ml * 32 + kh * 8];
            v16bf pa = pack_ab(*(const uint4*)pp, *(const uint4*)(pp + 16));
            asm volatile("" ::: "memory");
#pragma unroll
            for (int dt = 0; dt < 4; ++dt) {
                const bf16* vr = &vbuf[par][(dt * 16 + ml) * 32 + kh * 8];
                v16bf vbv = pack_ab(*(const uint4*)vr, *(const uint4*)(vr + 16));
                oacc[dt] = wmma_bf16(pa, vbv, oacc[dt]);
            }
        }
    }
    // epilogue: divide by softmax denominator, scatter to [B,S,NH*HD]
#pragma unroll
    for (int r = 0; r < 8; ++r) {
        float inv = 1.0f / li[r];
        int m = r + kh * 8;
        int s = qt * 16 + m;
        bf16* yp = Y + ((size_t)(b * SLEN + s)) * DIMSZ + h * HDIM;
#pragma unroll
        for (int dt = 0; dt < 4; ++dt)
            yp[dt * 16 + ml] = (bf16)(oacc[dt][r] * inv);
    }
}

extern "C" void kernel_launch(void* const* d_in, const int* in_sizes, int n_in,
                              void* d_out, int out_size, void* d_ws, size_t ws_size,
                              hipStream_t stream) {
    (void)in_sizes; (void)n_in; (void)out_size; (void)ws_size;
    const float* x  = (const float*)d_in[0];
    const float* Wq = (const float*)d_in[1];
    const float* Wk = (const float*)d_in[2];
    const float* Wv = (const float*)d_in[3];
    const float* Wo = (const float*)d_in[4];
    const float* qg = (const float*)d_in[5];
    float* out = (float*)d_out;

    char* w = (char*)d_ws;
    auto take = [&](size_t bytes) {
        char* p = w;
        w += (bytes + 255) & ~(size_t)255;
        return p;
    };
    const size_t KVD = (size_t)NKV * HDIM; // 256
    bf16* xb   = (bf16*)take((size_t)MTOK * DIMSZ * 2);
    bf16* wqb  = (bf16*)take((size_t)DIMSZ * DIMSZ * 2);
    bf16* wkb  = (bf16*)take(KVD * DIMSZ * 2);
    bf16* wvb  = (bf16*)take(KVD * DIMSZ * 2);
    bf16* wob  = (bf16*)take((size_t)DIMSZ * DIMSZ * 2);
    float* qf  = (float*)take((size_t)MTOK * DIMSZ * 4);
    float* kf  = (float*)take((size_t)MTOK * KVD * 4);
    float* vf  = (float*)take((size_t)MTOK * KVD * 4);
    bf16* qbh  = (bf16*)take((size_t)MTOK * DIMSZ * 2);
    bf16* kbh  = (bf16*)take((size_t)MTOK * KVD * 2);
    bf16* vtb  = (bf16*)take((size_t)MTOK * KVD * 2);
    bf16* yb   = (bf16*)take((size_t)MTOK * DIMSZ * 2);

    // stage bf16 activations/weights
    tobf16<<<(MTOK * DIMSZ + 255) / 256, 256, 0, stream>>>(x, xb, MTOK * DIMSZ);
    tobf16<<<(DIMSZ * DIMSZ + 255) / 256, 256, 0, stream>>>(Wq, wqb, DIMSZ * DIMSZ);
    tobf16<<<((int)KVD * DIMSZ + 255) / 256, 256, 0, stream>>>(Wk, wkb, (int)KVD * DIMSZ);
    tobf16<<<((int)KVD * DIMSZ + 255) / 256, 256, 0, stream>>>(Wv, wvb, (int)KVD * DIMSZ);
    tobf16<<<(DIMSZ * DIMSZ + 255) / 256, 256, 0, stream>>>(Wo, wob, DIMSZ * DIMSZ);

    // QKV projections (NT GEMMs via WMMA)
    gemm_nt_bf16<<<dim3(DIMSZ / 128, MTOK / 128), 128, 0, stream>>>(xb, wqb, qf, MTOK, DIMSZ, DIMSZ);
    gemm_nt_bf16<<<dim3((int)KVD / 128, MTOK / 128), 128, 0, stream>>>(xb, wkb, kf, MTOK, (int)KVD, DIMSZ);
    gemm_nt_bf16<<<dim3((int)KVD / 128, MTOK / 128), 128, 0, stream>>>(xb, wvb, vf, MTOK, (int)KVD, DIMSZ);

    // per-head RMSNorm + RoPE (+gain for q), layout shuffle to [B,heads,S,HD]
    normrope<<<(BATCH * NH * SLEN) / 8, 256, 0, stream>>>(qf, qbh, qg, NH);
    normrope<<<(BATCH * NKV * SLEN) / 8, 256, 0, stream>>>(kf, kbh, nullptr, NKV);
    vtrans<<<(MTOK * (int)KVD + 255) / 256, 256, 0, stream>>>(vf, vtb);

    // causal GQA flash attention (async LDS staging + WMMA)
    attn_fa<<<(BATCH * NH * (SLEN / 16)) / 8, 256, 0, stream>>>(qbh, kbh, vtb, yb);

    // output projection -> fp32 d_out
    gemm_nt_bf16<<<dim3(DIMSZ / 128, MTOK / 128), 128, 0, stream>>>(yb, wob, out, MTOK, DIMSZ, DIMSZ);
}
